// QIGAT_Corrected_39608188404034
// MI455X (gfx1250) — matrix operand order repaired and checked by reading, not compiled, not run on hardware
//
#include <hip/hip_runtime.h>
#include <hip/hip_bf16.h>

// ---------------------------------------------------------------------------
// QI-GAT forward for MI455X (gfx1250, wave32).
// GEMMs in bf16 via v_wmma_f32_16x16x32_bf16 with ASYNC global->LDS staging
// (global_load_async_to_lds_b128 + s_wait_asynccnt, 2-stage pipeline,
// double-buffered LDS, peeled epilogue); attention softmax in fp32 with
// atomic segment ops.
// ---------------------------------------------------------------------------

typedef __attribute__((ext_vector_type(16))) __bf16          v16bf;
typedef __attribute__((ext_vector_type(16))) unsigned short  v16u;
typedef __attribute__((ext_vector_type(8)))  float           v8f;

#define H_HEADS 4
#define CH      128
#define D1      512
#define DQ      256
#define F_IN    182
#define F_PAD   192   // 182 padded to multiple of 32

static __device__ __forceinline__ unsigned short f32_to_bf16(float f) {
    unsigned u = __builtin_bit_cast(unsigned, f);
    unsigned r = u + 0x7FFFu + ((u >> 16) & 1u);   // round-to-nearest-even
    return (unsigned short)(r >> 16);
}

// Monotonic uint key for float atomicMax (total order, works for negatives).
static __device__ __forceinline__ unsigned fkey(float f) {
    unsigned u = __builtin_bit_cast(unsigned, f);
    return (u & 0x80000000u) ? ~u : (u | 0x80000000u);
}
static __device__ __forceinline__ float fdecode(unsigned k) {
    unsigned u = (k & 0x80000000u) ? (k & 0x7FFFFFFFu) : ~k;
    return __builtin_bit_cast(float, u);
}

union FragU { v16u v; uint4 q[2]; };

// ---------------------------------------------------------------------------
// Tiled WMMA GEMM: C[M,Nc] = A[M,K](bf16, row-major) * BT[Nc,K](bf16,
// k-contiguous "W-layout") (+bias)(+relu).  K % 32 == 0, Nc % 64 == 0.
// Block = 256 threads = 8 waves (2 Mx4 N); block tile 64(M) x 64(N);
// each wave owns two 16x16 accumulators sharing one B fragment.
// Staging: async DMA global->LDS, double buffered, ASYNCcnt-pipelined,
// last iteration peeled so the hot loop is branch-free.
// ---------------------------------------------------------------------------
template <bool RELU>
__global__ __launch_bounds__(256) void wmma_gemm_bf16(
    const unsigned short* __restrict__ A,   // [M, K]
    const unsigned short* __restrict__ BT,  // [Nc, K]
    const float* __restrict__ bias,         // nullable
    float* __restrict__ Cf,                 // nullable
    unsigned short* __restrict__ Cb,        // nullable
    int M, int Nc, int K)
{
    __shared__ __align__(16) unsigned short As[2][64][40];  // 64 rows x 32 k, padded
    __shared__ __align__(16) unsigned short Bs[2][64][40];  // 64 n    x 32 k, padded

    const int tid  = threadIdx.x;
    const int lane = tid & 31;
    const int wave = tid >> 5;
    const int wm   = wave >> 2;                // 0..1
    const int wn   = wave & 3;                 // 0..3
    const int m0   = blockIdx.y * 64;
    const int n0   = blockIdx.x * 64;

    v8f acc0 = {}, acc1 = {};

    const int st_r = tid >> 2;                 // 0..63
    const int st_c = (tid & 3) * 8;            // 8 ushorts = 16B per thread

    // clamped row: tail rows read valid (unused) data -> branch-free staging
    const int marow = (m0 + st_r < M) ? (m0 + st_r) : (M - 1);
    const unsigned short* aRow = A  + (size_t)marow * K + st_c;
    const unsigned short* bRow = BT + (size_t)(n0 + st_r) * K + st_c;

    // LDS byte offsets of this thread's staging slots (both buffers)
    const unsigned ldsA0 = (unsigned)(uintptr_t)&As[0][st_r][st_c];
    const unsigned ldsA1 = (unsigned)(uintptr_t)&As[1][st_r][st_c];
    const unsigned ldsB0 = (unsigned)(uintptr_t)&Bs[0][st_r][st_c];
    const unsigned ldsB1 = (unsigned)(uintptr_t)&Bs[1][st_r][st_c];

    // async DMA: 16B/lane global -> LDS, tracked by ASYNCcnt (2 ops / tile)
    auto issue_tile = [](unsigned la, unsigned lb,
                         const unsigned short* ga, const unsigned short* gb) {
        asm volatile("global_load_async_to_lds_b128 %0, %1, off"
                     :: "v"(la), "v"(ga) : "memory");
        asm volatile("global_load_async_to_lds_b128 %0, %1, off"
                     :: "v"(lb), "v"(gb) : "memory");
    };

    const int mf   = (wm << 4) + (lane & 15);
    const int nf   = (wn << 4) + (lane & 15);
    const int kb_a = (lane < 16) ? 0 : 8;      // A frag: k pairs at kb, kb+16
    const int kb_b = (lane < 16) ? 0 : 8;      // B frag: k pairs at kb, kb+16

    auto compute = [&](int pp) {
        const unsigned short (*Ap)[40] = As[pp];
        const unsigned short (*Bp)[40] = Bs[pp];
        FragU a0, a1, b;
        a0.q[0] = *(const uint4*)&Ap[mf][kb_a];
        a0.q[1] = *(const uint4*)&Ap[mf][kb_a + 16];
        a1.q[0] = *(const uint4*)&Ap[mf + 32][kb_a];
        a1.q[1] = *(const uint4*)&Ap[mf + 32][kb_a + 16];
        b.q[0]  = *(const uint4*)&Bp[nf][kb_b];
        b.q[1]  = *(const uint4*)&Bp[nf][kb_b + 16];
        acc0 = __builtin_amdgcn_wmma_f32_16x16x32_bf16(
            false, __builtin_bit_cast(v16bf, a0.v),
            false, __builtin_bit_cast(v16bf, b.v),
            (short)0, acc0, false, false);
        acc1 = __builtin_amdgcn_wmma_f32_16x16x32_bf16(
            false, __builtin_bit_cast(v16bf, a1.v),
            false, __builtin_bit_cast(v16bf, b.v),
            (short)0, acc1, false, false);
    };

    issue_tile(ldsA0, ldsB0, aRow, bRow);      // prologue: stage tile 0

    const unsigned short* aNext = aRow + 32;   // running pointers, +32 elems/tile
    const unsigned short* bNext = bRow + 32;
    int p = 0;
    // steady state: branch-free — issue next, wait current, compute current
    for (int kt = 0; kt < K - 32; kt += 32, p ^= 1) {
        issue_tile(p ? ldsA0 : ldsA1, p ? ldsB0 : ldsB1, aNext, bNext);
        asm volatile("s_wait_asynccnt 0x2" ::: "memory");  // tile p landed
        __builtin_prefetch(aNext + 32, 0, 3);              // warm L2 for tile+2
        __builtin_prefetch(bNext + 32, 0, 3);
        aNext += 32; bNext += 32;
        __syncthreads();                       // tile p visible to all waves
        compute(p);
        __syncthreads();                       // all reads of tile p done
    }
    // epilogue: last tile
    asm volatile("s_wait_asynccnt 0x0" ::: "memory");
    __syncthreads();
    compute(p);

    // C layout: VGPR r -> M = r (+8 for lanes 16..31), N = lane&15
    const int n  = n0 + (wn << 4) + (lane & 15);
    const int mb = m0 + (wm << 4) + ((lane < 16) ? 0 : 8);
    const float bv = bias ? bias[n] : 0.0f;
    #pragma unroll
    for (int r = 0; r < 8; ++r) {
        int m = mb + r;
        if (m < M) {
            float v = acc0[r] + bv;
            if (RELU) v = v > 0.0f ? v : 0.0f;
            if (Cf) Cf[(size_t)m * Nc + n] = v;
            if (Cb) Cb[(size_t)m * Nc + n] = f32_to_bf16(v);
        }
        m += 32;
        if (m < M) {
            float v = acc1[r] + bv;
            if (RELU) v = v > 0.0f ? v : 0.0f;
            if (Cf) Cf[(size_t)m * Nc + n] = v;
            if (Cb) Cb[(size_t)m * Nc + n] = f32_to_bf16(v);
        }
    }
}

// ---------------------------------------------------------------------------
// Weight convert: BT[n*Kpad+k] = bf16(W[n*K+k]), zero-padded to Kpad cols.
// (out = x @ W.T  ->  BT is just W itself in bf16, k-contiguous.)
// ---------------------------------------------------------------------------
__global__ void cvt_weight_pad(const float* __restrict__ W, unsigned short* __restrict__ BT,
                               int K, int Kpad, int Nc)
{
    long long t = (long long)blockIdx.x * blockDim.x + threadIdx.x;
    long long total = (long long)Kpad * Nc;
    if (t >= total) return;
    int n = (int)(t / Kpad), k = (int)(t % Kpad);
    float v = (k < K) ? W[(size_t)n * K + k] : 0.0f;
    BT[t] = f32_to_bf16(v);
}

__global__ void cvt_x_pad(const float* __restrict__ x, unsigned short* __restrict__ A, int N)
{
    long long t = (long long)blockIdx.x * blockDim.x + threadIdx.x;
    long long total = (long long)N * F_PAD;
    if (t >= total) return;
    int n = (int)(t / F_PAD), k = (int)(t % F_PAD);
    float v = (k < F_IN) ? x[(size_t)n * F_IN + k] : 0.0f;
    A[t] = f32_to_bf16(v);
}

// ---------------------------------------------------------------------------
// Per-(node,head) attention logits: alS/alD[n,h] = sum_c P[n,h,c]*a[h,c]
// One wave per (n,h).
// ---------------------------------------------------------------------------
__global__ __launch_bounds__(256) void gat_scores(
    const float* __restrict__ P, const float* __restrict__ aS, const float* __restrict__ aD,
    float* __restrict__ alS, float* __restrict__ alD, int N)
{
    int gw = blockIdx.x * 8 + (threadIdx.x >> 5);
    if (gw >= N * H_HEADS) return;
    int lane = threadIdx.x & 31;
    int n = gw >> 2, h = gw & 3;
    const float* p  = P + (size_t)n * D1 + h * CH;
    const float* as = aS + h * CH;
    const float* ad = aD + h * CH;
    float ss = 0.f, sd = 0.f;
    #pragma unroll
    for (int j = 0; j < 4; ++j) {
        float v = p[lane + 32 * j];
        ss += v * as[lane + 32 * j];
        sd += v * ad[lane + 32 * j];
    }
    #pragma unroll
    for (int o = 16; o > 0; o >>= 1) {
        ss += __shfl_down(ss, o, 32);
        sd += __shfl_down(sd, o, 32);
    }
    if (lane == 0) { alS[gw] = ss; alD[gw] = sd; }
}

__global__ void init_nh(unsigned* __restrict__ emaxKey, float* __restrict__ denom, int N)
{
    int t = blockIdx.x * blockDim.x + threadIdx.x;
    if (t >= N * H_HEADS) return;
    emaxKey[t] = 0u;      // encodes "most negative"
    denom[t]   = 0.0f;
}

__global__ void init_out_bias(float* __restrict__ OUT, const float* __restrict__ bias, int N)
{
    long long t = (long long)blockIdx.x * blockDim.x + threadIdx.x;
    if (t >= (long long)N * D1) return;
    OUT[t] = bias[(int)(t & (D1 - 1))];
}

// edge pass 1: e = leaky_relu(alS[src]+alD[dst]); segment max via key atomicMax
__global__ void edge_pass1(const int* __restrict__ ei,
                           const float* __restrict__ alS, const float* __restrict__ alD,
                           float* __restrict__ ebuf, unsigned* __restrict__ emaxKey,
                           int E, int N)
{
    long long t = (long long)blockIdx.x * blockDim.x + threadIdx.x;
    long long total = (long long)(E + N) * H_HEADS;
    if (t >= total) return;
    int idx = (int)(t >> 2), h = (int)(t & 3);
    int s, d;
    if (idx < E) { s = ei[idx]; d = ei[E + idx]; } else { s = d = idx - E; }
    float e = alS[s * H_HEADS + h] + alD[d * H_HEADS + h];
    e = e > 0.0f ? e : 0.2f * e;
    ebuf[t] = e;
    atomicMax(&emaxKey[d * H_HEADS + h], fkey(e));
}

// edge pass 2: ex = exp(e - max); segment sum via float atomicAdd
__global__ void edge_pass2(const int* __restrict__ ei,
                           float* __restrict__ ebuf, const unsigned* __restrict__ emaxKey,
                           float* __restrict__ denom, int E, int N)
{
    long long t = (long long)blockIdx.x * blockDim.x + threadIdx.x;
    long long total = (long long)(E + N) * H_HEADS;
    if (t >= total) return;
    int idx = (int)(t >> 2), h = (int)(t & 3);
    int d = (idx < E) ? ei[E + idx] : (idx - E);
    float ex = __expf(ebuf[t] - fdecode(emaxKey[d * H_HEADS + h]));
    ebuf[t] = ex;
    atomicAdd(&denom[d * H_HEADS + h], ex);
}

// edge pass 3: OUT[dst,h,:] += P[src,h,:] * (ex/denom). One wave per (edge,head).
__global__ __launch_bounds__(256) void edge_aggregate(
    const int* __restrict__ ei, const float* __restrict__ P,
    const float* __restrict__ ebuf, const float* __restrict__ denom,
    float* __restrict__ OUT, int E, int N)
{
    long long gw = (long long)blockIdx.x * 8 + (threadIdx.x >> 5);
    long long total = (long long)(E + N) * H_HEADS;
    if (gw >= total) return;
    int lane = threadIdx.x & 31;
    int idx = (int)(gw >> 2), h = (int)(gw & 3);
    int s, d;
    if (idx < E) { s = ei[idx]; d = ei[E + idx]; } else { s = d = idx - E; }
    float alpha = ebuf[gw] / denom[d * H_HEADS + h];
    const float* ps = P   + (size_t)s * D1 + h * CH;
    float*       od = OUT + (size_t)d * D1 + h * CH;
    #pragma unroll
    for (int j = 0; j < 4; ++j)
        atomicAdd(&od[lane + 32 * j], ps[lane + 32 * j] * alpha);
}

// ---------------------------------------------------------------------------
// Row LayerNorm (+optional ELU), optional f32 and bf16 outputs. 128 thr/row.
// ---------------------------------------------------------------------------
__global__ __launch_bounds__(128) void layernorm_kernel(
    const float* __restrict__ X, const float* __restrict__ g, const float* __restrict__ be,
    float* __restrict__ Yf, unsigned short* __restrict__ Yb, int D, int elu)
{
    __shared__ float s1[4], s2[4];
    const int row = blockIdx.x, tid = threadIdx.x;
    const int lane = tid & 31, w = tid >> 5;
    const float* x = X + (size_t)row * D;
    float sum = 0.f, sq = 0.f;
    for (int j = tid; j < D; j += 128) { float v = x[j]; sum += v; sq += v * v; }
    #pragma unroll
    for (int o = 16; o > 0; o >>= 1) { sum += __shfl_down(sum, o, 32); sq += __shfl_down(sq, o, 32); }
    if (lane == 0) { s1[w] = sum; s2[w] = sq; }
    __syncthreads();
    if (tid == 0) {
        float a = 0.f, b = 0.f;
        #pragma unroll
        for (int i = 0; i < 4; ++i) { a += s1[i]; b += s2[i]; }
        s1[0] = a; s2[0] = b;
    }
    __syncthreads();
    const float mean = s1[0] / D;
    const float var  = s2[0] / D - mean * mean;
    const float rstd = rsqrtf(var + 1e-5f);
    for (int j = tid; j < D; j += 128) {
        float v = (x[j] - mean) * rstd * g[j] + be[j];
        if (elu) v = v > 0.f ? v : expm1f(v);
        if (Yf) Yf[(size_t)row * D + j] = v;
        if (Yb) Yb[(size_t)row * D + j] = f32_to_bf16(v);
    }
}

// phi = pi*tanh(z); HQ[n] = [cos(phi), sin(phi)] in bf16 ([N,1024])
__global__ void phase_kernel(const float* __restrict__ Z, unsigned short* __restrict__ HQ, int N)
{
    long long t = (long long)blockIdx.x * blockDim.x + threadIdx.x;
    if (t >= (long long)N * D1) return;
    int n = (int)(t >> 9), j = (int)(t & (D1 - 1));
    float phi = 3.14159265358979323846f * tanhf(Z[t]);
    float s, c;
    __sincosf(phi, &s, &c);
    HQ[(size_t)n * (2 * D1) + j]      = f32_to_bf16(c);
    HQ[(size_t)n * (2 * D1) + D1 + j] = f32_to_bf16(s);
}

// h_comb = R + res_scale * LN(hq)  ->  bf16 [N,256]
__global__ void combine_kernel(const float* __restrict__ R, const float* __restrict__ HQln,
                               const float* __restrict__ res_scale,
                               unsigned short* __restrict__ OUTb, int N)
{
    long long t = (long long)blockIdx.x * blockDim.x + threadIdx.x;
    if (t >= (long long)N * DQ) return;
    OUTb[t] = f32_to_bf16(R[t] + res_scale[0] * HQln[t]);
}

// out[n,{0,1}] = hid[n,:] . Wm2[{0,1},:] + bm2. One wave per node.
__global__ __launch_bounds__(256) void classifier_out(
    const float* __restrict__ hid, const float* __restrict__ Wm2,
    const float* __restrict__ bm2, float* __restrict__ out, int N)
{
    int gw = blockIdx.x * 8 + (threadIdx.x >> 5);
    if (gw >= N) return;
    int lane = threadIdx.x & 31;
    const float* h = hid + (size_t)gw * D1;
    float s0 = 0.f, s1 = 0.f;
    for (int j = lane; j < D1; j += 32) {
        float v = h[j];
        s0 += v * Wm2[j];
        s1 += v * Wm2[D1 + j];
    }
    #pragma unroll
    for (int o = 16; o > 0; o >>= 1) {
        s0 += __shfl_down(s0, o, 32);
        s1 += __shfl_down(s1, o, 32);
    }
    if (lane == 0) {
        out[(size_t)gw * 2]     = s0 + bm2[0];
        out[(size_t)gw * 2 + 1] = s1 + bm2[1];
    }
}

// ---------------------------------------------------------------------------
static inline int cdiv(long long a, long long b) { return (int)((a + b - 1) / b); }

extern "C" void kernel_launch(void* const* d_in, const int* in_sizes, int n_in,
                              void* d_out, int out_size, void* d_ws, size_t ws_size,
                              hipStream_t stream)
{
    const float* x       = (const float*)d_in[0];
    const int*   ei      = (const int*)  d_in[1];
    const float* W1      = (const float*)d_in[2];
    const float* a_src1  = (const float*)d_in[3];
    const float* a_dst1  = (const float*)d_in[4];
    const float* b1      = (const float*)d_in[5];
    const float* g1      = (const float*)d_in[6];
    const float* be1     = (const float*)d_in[7];
    const float* Wp      = (const float*)d_in[8];
    const float* bp      = (const float*)d_in[9];
    const float* Wc      = (const float*)d_in[10];
    const float* bc      = (const float*)d_in[11];
    const float* gq      = (const float*)d_in[12];
    const float* bq      = (const float*)d_in[13];
    const float* Wr      = (const float*)d_in[14];
    const float* br      = (const float*)d_in[15];
    const float* res_sc  = (const float*)d_in[16];
    const float* W2      = (const float*)d_in[17];
    const float* a_src2  = (const float*)d_in[18];
    const float* a_dst2  = (const float*)d_in[19];
    const float* b2      = (const float*)d_in[20];
    const float* g2      = (const float*)d_in[21];
    const float* be2     = (const float*)d_in[22];
    const float* Wm1     = (const float*)d_in[23];
    const float* bm1     = (const float*)d_in[24];
    const float* Wm2     = (const float*)d_in[25];
    const float* bm2     = (const float*)d_in[26];

    const int N = in_sizes[0] / F_IN;
    const int E = in_sizes[1] / 2;
    const long long EN = (long long)E + N;

    // ---- workspace bump allocator -------------------------------------
    char* ws = (char*)d_ws;
    size_t off = 0;
    auto alloc = [&](size_t bytes) -> void* {
        void* p = ws + off;
        off = (off + bytes + 255) & ~(size_t)255;
        return p;
    };
    unsigned short* ABF0 = (unsigned short*)alloc((size_t)N * 2 * D1 * 2); // x_pad / hq / h_comb
    unsigned short* ABF1 = (unsigned short*)alloc((size_t)N * D1 * 2);     // h1_bf / h2_bf
    float* P   = (float*)alloc((size_t)N * D1 * 4);                        // gemm outputs
    float* OUT = (float*)alloc((size_t)N * D1 * 4);                        // aggregation / h1 / h2
    float* R   = (float*)alloc((size_t)N * DQ * 4);                        // residual proj
    float* alS = (float*)alloc((size_t)N * H_HEADS * 4);
    float* alD = (float*)alloc((size_t)N * H_HEADS * 4);
    unsigned* emaxKey = (unsigned*)alloc((size_t)N * H_HEADS * 4);
    float* denom = (float*)alloc((size_t)N * H_HEADS * 4);
    float* ebuf  = (float*)alloc((size_t)EN * H_HEADS * 4);
    unsigned short* Bw1  = (unsigned short*)alloc((size_t)D1 * F_PAD * 2);
    unsigned short* Bwp  = (unsigned short*)alloc((size_t)D1 * D1 * 2);
    unsigned short* Bwc  = (unsigned short*)alloc((size_t)DQ * (2 * D1) * 2);
    unsigned short* Bwr  = (unsigned short*)alloc((size_t)DQ * D1 * 2);
    unsigned short* Bw2  = (unsigned short*)alloc((size_t)D1 * DQ * 2);
    unsigned short* Bwm1 = (unsigned short*)alloc((size_t)D1 * D1 * 2);
    (void)ws_size;

    const dim3 blk256(256), blk128(128);

    // ---- weight conversion (k-contiguous BT = bf16(W), zero-padded K) --
    cvt_weight_pad<<<cdiv((long long)D1 * F_PAD, 256), blk256, 0, stream>>>(W1,  Bw1,  F_IN, F_PAD, D1);
    cvt_weight_pad<<<cdiv((long long)D1 * D1,    256), blk256, 0, stream>>>(Wp,  Bwp,  D1,   D1,    D1);
    cvt_weight_pad<<<cdiv((long long)DQ * 2*D1,  256), blk256, 0, stream>>>(Wc,  Bwc,  2*D1, 2*D1,  DQ);
    cvt_weight_pad<<<cdiv((long long)DQ * D1,    256), blk256, 0, stream>>>(Wr,  Bwr,  D1,   D1,    DQ);
    cvt_weight_pad<<<cdiv((long long)D1 * DQ,    256), blk256, 0, stream>>>(W2,  Bw2,  DQ,   DQ,    D1);
    cvt_weight_pad<<<cdiv((long long)D1 * D1,    256), blk256, 0, stream>>>(Wm1, Bwm1, D1,   D1,    D1);

    const int gmM = cdiv(N, 64);
    const dim3 g512(D1 / 64, gmM), g256(DQ / 64, gmM);
    const int nhBlocks  = cdiv((long long)N * H_HEADS, 8);       // wave-per-(n,h)
    const int edgeThr   = cdiv(EN * H_HEADS, 256);
    const int edgeWaves = cdiv(EN * H_HEADS, 8);

    // ---- GAT layer 1 ---------------------------------------------------
    cvt_x_pad<<<cdiv((long long)N * F_PAD, 256), blk256, 0, stream>>>(x, ABF0, N);
    wmma_gemm_bf16<false><<<g512, blk256, 0, stream>>>(ABF0, Bw1, nullptr, P, nullptr, N, D1, F_PAD);
    gat_scores<<<nhBlocks, blk256, 0, stream>>>(P, a_src1, a_dst1, alS, alD, N);
    init_nh<<<cdiv((long long)N * H_HEADS, 256), blk256, 0, stream>>>(emaxKey, denom, N);
    edge_pass1<<<edgeThr, blk256, 0, stream>>>(ei, alS, alD, ebuf, emaxKey, E, N);
    edge_pass2<<<edgeThr, blk256, 0, stream>>>(ei, ebuf, emaxKey, denom, E, N);
    init_out_bias<<<cdiv((long long)N * D1, 256), blk256, 0, stream>>>(OUT, b1, N);
    edge_aggregate<<<edgeWaves, blk256, 0, stream>>>(ei, P, ebuf, denom, OUT, E, N);
    layernorm_kernel<<<N, blk128, 0, stream>>>(OUT, g1, be1, OUT, ABF1, D1, /*elu=*/1); // h1

    // ---- quantum phase block ------------------------------------------
    wmma_gemm_bf16<false><<<g512, blk256, 0, stream>>>(ABF1, Bwp, bp, P, nullptr, N, D1, D1);   // z
    wmma_gemm_bf16<false><<<g256, blk256, 0, stream>>>(ABF1, Bwr, br, R, nullptr, N, DQ, D1);   // residual
    phase_kernel<<<cdiv((long long)N * D1, 256), blk256, 0, stream>>>(P, ABF0, N);              // hq bf16
    wmma_gemm_bf16<false><<<g256, blk256, 0, stream>>>(ABF0, Bwc, bc, P, nullptr, N, DQ, 2*D1); // hqc
    layernorm_kernel<<<N, blk128, 0, stream>>>(P, gq, bq, P, nullptr, DQ, /*elu=*/0);
    combine_kernel<<<cdiv((long long)N * DQ, 256), blk256, 0, stream>>>(R, P, res_sc, ABF0, N);

    // ---- GAT layer 2 ---------------------------------------------------
    wmma_gemm_bf16<false><<<g512, blk256, 0, stream>>>(ABF0, Bw2, nullptr, P, nullptr, N, D1, DQ);
    gat_scores<<<nhBlocks, blk256, 0, stream>>>(P, a_src2, a_dst2, alS, alD, N);
    init_nh<<<cdiv((long long)N * H_HEADS, 256), blk256, 0, stream>>>(emaxKey, denom, N);
    edge_pass1<<<edgeThr, blk256, 0, stream>>>(ei, alS, alD, ebuf, emaxKey, E, N);
    edge_pass2<<<edgeThr, blk256, 0, stream>>>(ei, ebuf, emaxKey, denom, E, N);
    init_out_bias<<<cdiv((long long)N * D1, 256), blk256, 0, stream>>>(OUT, b2, N);
    edge_aggregate<<<edgeWaves, blk256, 0, stream>>>(ei, P, ebuf, denom, OUT, E, N);
    layernorm_kernel<<<N, blk128, 0, stream>>>(OUT, g2, be2, OUT, ABF1, D1, /*elu=*/1); // h2

    // ---- classifier ----------------------------------------------------
    wmma_gemm_bf16<true><<<g512, blk256, 0, stream>>>(ABF1, Bwm1, bm1, P, nullptr, N, D1, D1); // relu
    classifier_out<<<cdiv(N, 8), blk256, 0, stream>>>(P, Wm2, bm2, (float*)d_out, N);
}
